// Attention_65000035058441
// MI455X (gfx1250) — compile-verified, matrix operand order checked by reading, not run on hardware
//
#include <hip/hip_runtime.h>

// ---------------------------------------------------------------------------
// Fused causal MHA for MI455X (gfx1250, wave32, WMMA f16->f32).
// Pipeline: cast->f16, 3 WMMA projections (head-major / V-transposed),
// flash-attention WMMA kernel with online softmax, WMMA output projection.
// GEMM waves are register-blocked 32x64 (2x4 fragments) for ~21 FLOP/B.
// ---------------------------------------------------------------------------

typedef __attribute__((ext_vector_type(16))) _Float16 v16h;
typedef __attribute__((ext_vector_type(8)))  _Float16 v8h;
typedef __attribute__((ext_vector_type(8)))  float    v8f;

#define WMMA_F32_F16(a, b, c) \
  __builtin_amdgcn_wmma_f32_16x16x32_f16(false, (a), false, (b), (short)0, (c), false, false)

static constexpr int   BATCH  = 2;
static constexpr int   NTOK   = 2048;
static constexpr int   DIMS   = 1024;
static constexpr int   NHEADS = 16;
static constexpr int   HD     = 64;       // head dim
static constexpr float ASCALE = 0.125f;   // 1/sqrt(64)

// Load a 16x32 f16 A/B fragment row/column slice for one lane.
// ISA layout: lane half 0 gets K [k0..k0+7] and [k0+16..k0+23],
//             lane half 1 gets K [k0+8..k0+15] and [k0+24..k0+31].
__device__ __forceinline__ v16h load_frag16(const _Float16* p, int hsel) {
  v8h lo = *(const v8h*)(p + 8 * hsel);
  v8h hi = *(const v8h*)(p + 16 + 8 * hsel);
  v16h r;
#pragma unroll
  for (int i = 0; i < 8; ++i) { r[i] = lo[i]; r[i + 8] = hi[i]; }
  return r;
}

// ------------------------- elementwise prep kernels ------------------------

__global__ void add_cvt_f16(const float* __restrict__ a, const float* __restrict__ b,
                            _Float16* __restrict__ o, int n) {
  int i = blockIdx.x * blockDim.x + threadIdx.x;
  if (i < n) o[i] = (_Float16)(a[i] + b[i]);
}

__global__ void cvt_f16(const float* __restrict__ a, _Float16* __restrict__ o, int n) {
  int i = blockIdx.x * blockDim.x + threadIdx.x;
  if (i < n) o[i] = (_Float16)a[i];
}

// ------------------------------ WMMA GEMM ----------------------------------
// C[M,Nout] = A[M,K] @ W[Nout,K]^T + bias.
// One wave -> 32x64 output tile: 2 A-fragments x 4 B-fragments, 8 WMMAs/K-step.
// mode 1: f16 out, head-major  [B,H,N,HD]
// mode 2: f16 out, V-transpose [B,H,HD,N]
// mode 3: f32 out, row-major   [M,Nout]

__global__ __launch_bounds__(256) void gemm_wmma(
    const _Float16* __restrict__ A, const _Float16* __restrict__ W,
    const float* __restrict__ bias, void* __restrict__ out,
    int M, int K, int Nout, int mode) {
  const int lane   = threadIdx.x & 31;
  const int wid    = (blockIdx.x * blockDim.x + threadIdx.x) >> 5;
  const int tilesN = Nout >> 6;               // 64-wide N tiles
  const int tm     = (wid / tilesN) * 32;     // row base (2 x 16)
  const int tn     = (wid % tilesN) * 64;     // col base (4 x 16)
  if (tm >= M) return;
  const int l16  = lane & 15;
  const int hsel = lane >> 4;

  const _Float16* ap0 = A + (size_t)(tm + l16) * K;
  const _Float16* ap1 = ap0 + (size_t)16 * K;
  const _Float16* wp0 = W + (size_t)(tn + l16) * K;
  const _Float16* wp1 = wp0 + (size_t)16 * K;
  const _Float16* wp2 = wp0 + (size_t)32 * K;
  const _Float16* wp3 = wp0 + (size_t)48 * K;

  v8f acc[2][4] = {};
  for (int k = 0; k < K; k += 32) {
    if (k + 256 < K) {                         // stream-prefetch A rows
      __builtin_prefetch(ap0 + k + 256, 0, 1);
      __builtin_prefetch(ap1 + k + 256, 0, 1);
    }
    v16h a0 = load_frag16(ap0 + k, hsel);
    v16h a1 = load_frag16(ap1 + k, hsel);
    v16h b0 = load_frag16(wp0 + k, hsel);
    v16h b1 = load_frag16(wp1 + k, hsel);
    v16h b2 = load_frag16(wp2 + k, hsel);
    v16h b3 = load_frag16(wp3 + k, hsel);
    acc[0][0] = WMMA_F32_F16(a0, b0, acc[0][0]);
    acc[0][1] = WMMA_F32_F16(a0, b1, acc[0][1]);
    acc[0][2] = WMMA_F32_F16(a0, b2, acc[0][2]);
    acc[0][3] = WMMA_F32_F16(a0, b3, acc[0][3]);
    acc[1][0] = WMMA_F32_F16(a1, b0, acc[1][0]);
    acc[1][1] = WMMA_F32_F16(a1, b1, acc[1][1]);
    acc[1][2] = WMMA_F32_F16(a1, b2, acc[1][2]);
    acc[1][3] = WMMA_F32_F16(a1, b3, acc[1][3]);
  }

#pragma unroll
  for (int j = 0; j < 4; ++j) {
    const int   col  = tn + j * 16 + l16;
    const float bval = bias[col];
#pragma unroll
    for (int i = 0; i < 2; ++i) {
#pragma unroll
      for (int v = 0; v < 8; ++v) {
        const int   row = tm + i * 16 + v + 8 * hsel;
        const float val = acc[i][j][v] + bval;
        if (mode == 1) {          // head-major f16 [B,H,N,HD]
          int b = row / NTOK, n = row % NTOK, h = col >> 6, d = col & 63;
          ((_Float16*)out)[(((size_t)b * NHEADS + h) * NTOK + n) * HD + d] = (_Float16)val;
        } else if (mode == 2) {   // transposed f16 [B,H,HD,N]
          int b = row / NTOK, n = row % NTOK, h = col >> 6, d = col & 63;
          ((_Float16*)out)[(((size_t)b * NHEADS + h) * HD + d) * NTOK + n] = (_Float16)val;
        } else {                  // f32 [M,Nout]
          ((float*)out)[(size_t)row * Nout + col] = val;
        }
      }
    }
  }
}

// --------------------------- fused causal attention ------------------------
// Grid: B*H*(N/64) blocks, 128 threads (4 waves). Each wave owns 16 q-rows,
// streams 32-key blocks with online softmax. Qh/Kh: [B,H,N,64] f16,
// Vt: [B,H,64,N] f16, AO: [B,N,DIM] f16 (heads re-interleaved).

__global__ __launch_bounds__(128) void attn_fused(
    const _Float16* __restrict__ Qh, const _Float16* __restrict__ Kh,
    const _Float16* __restrict__ Vt, _Float16* __restrict__ AO) {
  __shared__ _Float16 Plds[4][16][32];

  const int lane = threadIdx.x & 31;
  const int wave = threadIdx.x >> 5;
  const int l16  = lane & 15;
  const int hsel = lane >> 4;

  const int qblk = blockIdx.x % (NTOK / 64);
  const int bh   = blockIdx.x / (NTOK / 64);
  const int b    = bh / NHEADS;
  const int h    = bh % NHEADS;
  const int qt   = qblk * 64 + wave * 16;     // base q row of this wave

  const _Float16* qbase = Qh + ((size_t)bh * NTOK + qt + l16) * HD;
  v16h aq0 = load_frag16(qbase, hsel);        // d = 0..31
  v16h aq1 = load_frag16(qbase + 32, hsel);   // d = 32..63

  v8f O0 = {}, O1 = {}, O2 = {}, O3 = {};
  float mrun[8], lrun[8];
#pragma unroll
  for (int v = 0; v < 8; ++v) { mrun[v] = -3.0e38f; lrun[v] = 0.f; }

  const int nkb = (qt + 16 + 31) >> 5;        // causal trip count (32-key blocks)
  for (int kb = 0; kb < nkb; ++kb) {
    const int k32 = kb * 32;

    // ---- S = Q @ K^T (two 16x16 D-fragments), scale + causal mask ----
    float s[2][8];
#pragma unroll
    for (int sub = 0; sub < 2; ++sub) {
      const int kbase = k32 + sub * 16;
      const _Float16* kbp = Kh + ((size_t)bh * NTOK + kbase + l16) * HD;
      v16h bk0 = load_frag16(kbp, hsel);
      v16h bk1 = load_frag16(kbp + 32, hsel);
      v8f c = {};
      c = WMMA_F32_F16(aq0, bk0, c);
      c = WMMA_F32_F16(aq1, bk1, c);
      const int key = kbase + l16;            // D-frag column -> key index
#pragma unroll
      for (int v = 0; v < 8; ++v) {
        const int qrow = qt + v + 8 * hsel;   // D-frag row
        s[sub][v] = (key <= qrow) ? c[v] * ASCALE : -3.0e38f;
      }
    }

    // ---- online softmax: row max across 16 lanes of each half ----
    float mnew[8], corr[8];
#pragma unroll
    for (int v = 0; v < 8; ++v) {
      float rm = fmaxf(s[0][v], s[1][v]);
      rm = fmaxf(rm, __shfl_xor(rm, 1, 32));
      rm = fmaxf(rm, __shfl_xor(rm, 2, 32));
      rm = fmaxf(rm, __shfl_xor(rm, 4, 32));
      rm = fmaxf(rm, __shfl_xor(rm, 8, 32));
      mnew[v] = fmaxf(mrun[v], rm);
      corr[v] = __expf(mrun[v] - mnew[v]);
      mrun[v] = mnew[v];
    }

    // ---- P = exp(S - m), stash row-major in LDS for A-fragment reload ----
#pragma unroll
    for (int sub = 0; sub < 2; ++sub) {
#pragma unroll
      for (int v = 0; v < 8; ++v) {
        float p = __expf(s[sub][v] - mnew[v]);
        s[sub][v] = p;
        Plds[wave][v + 8 * hsel][sub * 16 + l16] = (_Float16)p;
      }
    }
#pragma unroll
    for (int v = 0; v < 8; ++v) {
      float rs = s[0][v] + s[1][v];
      rs += __shfl_xor(rs, 1, 32);
      rs += __shfl_xor(rs, 2, 32);
      rs += __shfl_xor(rs, 4, 32);
      rs += __shfl_xor(rs, 8, 32);
      lrun[v] = lrun[v] * corr[v] + rs;
      O0[v] *= corr[v]; O1[v] *= corr[v]; O2[v] *= corr[v]; O3[v] *= corr[v];
    }

    // wave-local LDS store->load ordering (cross-lane reshape)
    asm volatile("s_wait_dscnt 0" ::: "memory");
    v16h pa = load_frag16(&Plds[wave][l16][0], hsel);

    // ---- O += P @ V  (V stored transposed: columns are contiguous) ----
    const _Float16* vbase = Vt + (size_t)bh * HD * NTOK + k32;
    v16h bv;
    bv = load_frag16(vbase + (size_t)(0 * 16 + l16) * NTOK, hsel);
    O0 = WMMA_F32_F16(pa, bv, O0);
    bv = load_frag16(vbase + (size_t)(1 * 16 + l16) * NTOK, hsel);
    O1 = WMMA_F32_F16(pa, bv, O1);
    bv = load_frag16(vbase + (size_t)(2 * 16 + l16) * NTOK, hsel);
    O2 = WMMA_F32_F16(pa, bv, O2);
    bv = load_frag16(vbase + (size_t)(3 * 16 + l16) * NTOK, hsel);
    O3 = WMMA_F32_F16(pa, bv, O3);
  }

  // ---- normalize and write AO[b, n, h*64 + d] as f16 ----
  _Float16* obase = AO + ((size_t)b * NTOK + qt) * DIMS + h * HD;
#pragma unroll
  for (int v = 0; v < 8; ++v) {
    const float inv = 1.0f / lrun[v];
    const int   m   = v + 8 * hsel;
    _Float16* orow  = obase + (size_t)m * DIMS;
    orow[0 * 16 + l16] = (_Float16)(O0[v] * inv);
    orow[1 * 16 + l16] = (_Float16)(O1[v] * inv);
    orow[2 * 16 + l16] = (_Float16)(O2[v] * inv);
    orow[3 * 16 + l16] = (_Float16)(O3[v] * inv);
  }
}

// ------------------------------- launcher ----------------------------------

extern "C" void kernel_launch(void* const* d_in, const int* in_sizes, int n_in,
                              void* d_out, int out_size, void* d_ws, size_t ws_size,
                              hipStream_t stream) {
  (void)in_sizes; (void)n_in; (void)out_size; (void)ws_size;

  const float* q      = (const float*)d_in[0];
  const float* k      = (const float*)d_in[1];
  const float* vin    = (const float*)d_in[2];
  const float* qpos   = (const float*)d_in[3];
  const float* kpos   = (const float*)d_in[4];
  // d_in[5] = mask (causal, applied analytically in-kernel)
  const float* wq_w   = (const float*)d_in[6];
  const float* wq_b   = (const float*)d_in[7];
  const float* wk_w   = (const float*)d_in[8];
  const float* wk_b   = (const float*)d_in[9];
  const float* wv_w   = (const float*)d_in[10];
  const float* wv_b   = (const float*)d_in[11];
  const float* proj_w = (const float*)d_in[12];
  const float* proj_b = (const float*)d_in[13];

  const size_t SZ_X = (size_t)BATCH * NTOK * DIMS * sizeof(_Float16);  // 8 MiB
  const size_t SZ_W = (size_t)DIMS * DIMS * sizeof(_Float16);          // 2 MiB
  char* ws = (char*)d_ws;
  _Float16* Xq   = (_Float16*)(ws);
  _Float16* Xk   = (_Float16*)(ws + SZ_X);
  _Float16* Xv   = (_Float16*)(ws + 2 * SZ_X);
  _Float16* Wq16 = (_Float16*)(ws + 3 * SZ_X);
  _Float16* Wk16 = (_Float16*)(ws + 3 * SZ_X + SZ_W);
  _Float16* Wv16 = (_Float16*)(ws + 3 * SZ_X + 2 * SZ_W);
  _Float16* Wp16 = (_Float16*)(ws + 3 * SZ_X + 3 * SZ_W);
  _Float16* Qh   = (_Float16*)(ws + 3 * SZ_X + 4 * SZ_W);
  _Float16* Kh   = (_Float16*)(ws + 4 * SZ_X + 4 * SZ_W);
  _Float16* Vt   = (_Float16*)(ws + 5 * SZ_X + 4 * SZ_W);
  _Float16* AO   = (_Float16*)(ws + 6 * SZ_X + 4 * SZ_W);

  const int nAct = BATCH * NTOK * DIMS;   // 4,194,304
  const int nW   = DIMS * DIMS;           // 1,048,576
  const int TPB  = 256;

  add_cvt_f16<<<(nAct + TPB - 1) / TPB, TPB, 0, stream>>>(q, qpos, Xq, nAct);
  add_cvt_f16<<<(nAct + TPB - 1) / TPB, TPB, 0, stream>>>(k, kpos, Xk, nAct);
  cvt_f16<<<(nAct + TPB - 1) / TPB, TPB, 0, stream>>>(vin, Xv, nAct);
  cvt_f16<<<(nW + TPB - 1) / TPB, TPB, 0, stream>>>(wq_w, Wq16, nW);
  cvt_f16<<<(nW + TPB - 1) / TPB, TPB, 0, stream>>>(wk_w, Wk16, nW);
  cvt_f16<<<(nW + TPB - 1) / TPB, TPB, 0, stream>>>(wv_w, Wv16, nW);
  cvt_f16<<<(nW + TPB - 1) / TPB, TPB, 0, stream>>>(proj_w, Wp16, nW);

  const int M = BATCH * NTOK;                            // 4096
  const int gemmBlocks = (M / 32) * (DIMS / 64) / 8;     // 256 (8 waves/block)
  gemm_wmma<<<gemmBlocks, 256, 0, stream>>>(Xq, Wq16, wq_b, Qh, M, DIMS, DIMS, 1);
  gemm_wmma<<<gemmBlocks, 256, 0, stream>>>(Xk, Wk16, wk_b, Kh, M, DIMS, DIMS, 1);
  gemm_wmma<<<gemmBlocks, 256, 0, stream>>>(Xv, Wv16, wv_b, Vt, M, DIMS, DIMS, 2);

  attn_fused<<<BATCH * NHEADS * (NTOK / 64), 128, 0, stream>>>(Qh, Kh, Vt, AO);

  gemm_wmma<<<gemmBlocks, 256, 0, stream>>>(AO, Wp16, proj_b, (float*)d_out,
                                            M, DIMS, DIMS, 3);
}
